// CausalSelfAttention_55207509622857
// MI455X (gfx1250) — matrix operand order repaired and checked by reading, not compiled, not run on hardware
//
#include <hip/hip_runtime.h>

#define B_ 4
#define T_ 2048
#define C_ 1024
#define H_ 16
#define D_ 64

typedef __attribute__((ext_vector_type(16))) _Float16 v16h;
typedef __attribute__((ext_vector_type(8)))  _Float16 v8h;
typedef __attribute__((ext_vector_type(8)))  float    v8f;
typedef __attribute__((ext_vector_type(4)))  unsigned int u32x4;
typedef __attribute__((ext_vector_type(8)))  int          i32x8;
typedef __attribute__((ext_vector_type(4)))  int          i32x4;

union V16 { v16h v; v8h h[2]; };

static __device__ __forceinline__ v8f wmma_f16(v16h a, v16h b, v8f c) {
  // D = A(16x32 f16) * B(32x16 f16) + C(16x16 f32)
  return __builtin_amdgcn_wmma_f32_16x16x32_f16(false, a, false, b, (short)0, c, false, false);
}

// ---------------------------------------------------------------------------
// Tiled GEMM: C[M,N] = A[M,K] @ B[K,N] + bias[N]; A,B fp32 row-major,
// converted to f16 in LDS, f32 accumulation via v_wmma_f32_16x16x32_f16.
// Block tile 64x64, K step 32; 8 waves, each wave computes 16x32.
// Staging: 2x global_load_b128 per thread per tile (single wait), then one
// ds_store_b128 (A) / b16 scatter (B transpose). Next K-step prefetched.
// ---------------------------------------------------------------------------
__global__ __launch_bounds__(256) void gemm_bias_f16(
    const float* __restrict__ A, const float* __restrict__ Bm,
    const float* __restrict__ bias, float* __restrict__ Cm,
    int M, int N, int K)
{
  __shared__ __align__(64) _Float16 As[64][32];   // [m][k]
  __shared__ __align__(64) _Float16 Bs[64][32];   // [n][k] (transposed)
  const int tid  = threadIdx.x;
  const int wave = tid >> 5, lane = tid & 31;
  const int half = lane >> 4, l15 = lane & 15;
  const int m0 = blockIdx.y * 64, n0 = blockIdx.x * 64;
  const int rowBase = (wave & 3) * 16;   // 4 M-subtiles
  const int colBase = (wave >> 2) * 32;  // 2 N-halves, 2 subtiles each

  // Per-thread staging coordinates
  const int ar = tid >> 2, ac = (tid & 3) << 3;   // A: row 0..63, col 0,8,16,24
  const int bk = tid >> 3, bn = (tid & 7) << 3;   // B: k 0..31, n 0,8,..,56

  v8f acc0 = {0.f,0.f,0.f,0.f,0.f,0.f,0.f,0.f};
  v8f acc1 = {0.f,0.f,0.f,0.f,0.f,0.f,0.f,0.f};

  for (int k0 = 0; k0 < K; k0 += 32) {
    __syncthreads();
    {
      // A: 8 contiguous floats -> one v8h LDS store (row-major)
      const float4* ap = (const float4*)&A[(size_t)(m0 + ar) * K + k0 + ac];
      float4 a0 = ap[0], a1 = ap[1];
      v8h ah;
      ah[0]=(_Float16)a0.x; ah[1]=(_Float16)a0.y; ah[2]=(_Float16)a0.z; ah[3]=(_Float16)a0.w;
      ah[4]=(_Float16)a1.x; ah[5]=(_Float16)a1.y; ah[6]=(_Float16)a1.z; ah[7]=(_Float16)a1.w;
      *(v8h*)&As[ar][ac] = ah;
      // B: 8 contiguous floats along N -> b16 scatter into transposed Bs
      const float4* bp = (const float4*)&Bm[(size_t)(k0 + bk) * N + n0 + bn];
      float4 b0v = bp[0], b1v = bp[1];
      Bs[bn + 0][bk] = (_Float16)b0v.x; Bs[bn + 1][bk] = (_Float16)b0v.y;
      Bs[bn + 2][bk] = (_Float16)b0v.z; Bs[bn + 3][bk] = (_Float16)b0v.w;
      Bs[bn + 4][bk] = (_Float16)b1v.x; Bs[bn + 5][bk] = (_Float16)b1v.y;
      Bs[bn + 6][bk] = (_Float16)b1v.z; Bs[bn + 7][bk] = (_Float16)b1v.w;
    }
    // Prefetch next K-step tiles into cache while this step computes.
    if (k0 + 32 < K) {
      __builtin_prefetch(&A[(size_t)(m0 + ar) * K + (k0 + 32) + ac], 0, 0);
      __builtin_prefetch(&Bm[(size_t)(k0 + 32 + bk) * N + n0 + bn], 0, 0);
    }
    __syncthreads();
    // A fragment: lane half 0 -> K 0..15; half 1 -> K 16..31
    V16 af;
    const _Float16* arow = &As[rowBase + l15][0];
    af.h[0] = *(const v8h*)(arow + half * 16);
    af.h[1] = *(const v8h*)(arow + half * 16 + 8);
    // B fragment: lane half 0 -> K 0..15; half 1 -> K 16..31 (contiguous in Bs)
    v16h b0 = *(const v16h*)(&Bs[colBase +      l15][0] + half * 16);
    v16h b1 = *(const v16h*)(&Bs[colBase + 16 + l15][0] + half * 16);
    acc0 = wmma_f16(af.v, b0, acc0);
    acc1 = wmma_f16(af.v, b1, acc1);
  }

  // C/D layout: vgpr i -> M = half*8 + i ; N = l15 within subtile
  const int gr0 = m0 + rowBase + half * 8;
  const int gc  = n0 + colBase + l15;
  #pragma unroll
  for (int i = 0; i < 8; ++i) {
    Cm[(size_t)(gr0 + i) * N + gc]      = acc0[i] + bias[gc];
    Cm[(size_t)(gr0 + i) * N + gc + 16] = acc1[i] + bias[gc + 16];
  }
}

// ---------------------------------------------------------------------------
// Post-QKV: split heads, lambda-blend V, RMSNorm + RoPE on Q/K.
// One wave32 per (b,h,t) row; lane j handles elements j and j+32 of D=64.
// ---------------------------------------------------------------------------
__global__ __launch_bounds__(256) void qkv_post(
    const float* __restrict__ qkv, const float* __restrict__ lambp,
    float* __restrict__ value_out,
    _Float16* __restrict__ Qf, _Float16* __restrict__ Kf, _Float16* __restrict__ Vf)
{
  const int wave = threadIdx.x >> 5, lane = threadIdx.x & 31;
  const int row = blockIdx.x * 8 + wave;          // 0 .. B*H*T-1
  const int t = row % T_;
  const int h = (row / T_) % H_;
  const int b = row / (T_ * H_);
  const float lamb = lambp[0];

  const float* base = qkv + (size_t)(b * T_ + t) * (3 * C_) + h * D_;
  float q1 = base[lane],          q2 = base[lane + 32];
  float k1 = base[C_ + lane],     k2 = base[C_ + lane + 32];
  float v1 = base[2*C_ + lane],   v2 = base[2*C_ + lane + 32];
  v1 = (1.f - lamb) * v1 + lamb * v1;
  v2 = (1.f - lamb) * v2 + lamb * v2;

  float sq = q1*q1 + q2*q2, sk = k1*k1 + k2*k2;
  #pragma unroll
  for (int m = 1; m < 32; m <<= 1) {
    sq += __shfl_xor(sq, m, 32);
    sk += __shfl_xor(sk, m, 32);
  }
  const float rq = rsqrtf(sq * (1.f/64.f) + 1e-6f);
  const float rk = rsqrtf(sk * (1.f/64.f) + 1e-6f);
  q1 *= rq; q2 *= rq; k1 *= rk; k2 *= rk;

  const float ang = (float)t * __powf(10000.f, -(float)lane * (1.f/32.f));
  const float cs = __cosf(ang), sn = __sinf(ang);
  const float qo1 =  q1*cs + q2*sn, qo2 = -q1*sn + q2*cs;
  const float ko1 =  k1*cs + k2*sn, ko2 = -k1*sn + k2*cs;

  const size_t o = ((size_t)(b * H_ + h) * T_ + t) * D_ + lane;
  Qf[o] = (_Float16)qo1; Qf[o + 32] = (_Float16)qo2;
  Kf[o] = (_Float16)ko1; Kf[o + 32] = (_Float16)ko2;
  Vf[o] = (_Float16)v1;  Vf[o + 32] = (_Float16)v2;
  value_out[o] = v1;     value_out[o + 32] = v2;
}

// ---------------------------------------------------------------------------
// Flash attention: grid (T/128, B*H), 8 waves; wave w owns q rows
// [q0, q0+16). KV tiles of 32. K tile staged once per block via the
// Tensor Data Mover (tensor_load_to_lds, TENSORcnt); V staged transposed
// (b128 global load + b16 LDS scatter). Online softmax in f32; P restaged
// through per-wave LDS to A-fragment layout. Causal.
// ---------------------------------------------------------------------------
__global__ __launch_bounds__(256) void flash_attn(
    const _Float16* __restrict__ Qf, const _Float16* __restrict__ Kf,
    const _Float16* __restrict__ Vf, float* __restrict__ attnO)
{
  __shared__ __align__(64) _Float16 Ks[32][64];    // [kv][d]  (TDM-staged)
  __shared__ __align__(64) _Float16 Vt[64][32];    // [d][kv]  (transposed)
  __shared__ __align__(64) _Float16 Ps[8][16][32]; // per-wave P tile
  const int tid  = threadIdx.x;
  const int wave = tid >> 5, lane = tid & 31;
  const int half = lane >> 4, l15 = lane & 15;
  const int bh = blockIdx.y;
  const int b = bh / H_, h = bh % H_;
  const size_t base = (size_t)bh * T_ * D_;
  const int q0 = blockIdx.x * 128 + wave * 16;

  // Q fragments (D=64 -> 2 chunks of K=32)
  V16 qf[2];
  #pragma unroll
  for (int c = 0; c < 2; ++c) {
    const _Float16* p = Qf + base + (size_t)(q0 + l15) * D_ + c * 32;
    qf[c].h[0] = *(const v8h*)(p + half * 16);
    qf[c].h[1] = *(const v8h*)(p + half * 16 + 8);
  }

  v8f o[4];
  float m8[8], l8[8];
  #pragma unroll
  for (int c2 = 0; c2 < 4; ++c2) {
    #pragma unroll
    for (int i = 0; i < 8; ++i) o[c2][i] = 0.f;
  }
  #pragma unroll
  for (int i = 0; i < 8; ++i) { m8[i] = -1e30f; l8[i] = 0.f; }

  const unsigned ldsKs = (unsigned)(unsigned long long)(&Ks[0][0]); // LDS byte offset
  const int vkv = tid >> 3, vdb = (tid & 7) << 3;  // V staging coords

  const int kvEnd = blockIdx.x * 128 + 128;
  for (int kv0 = 0; kv0 < kvEnd; kv0 += 32) {
    __syncthreads();
    // --- TDM: DMA K tile [kv0..kv0+32) x D into LDS (row-major) -----------
    if (wave == 0) {
      unsigned long long ga =
          (unsigned long long)(Kf + base + (size_t)kv0 * D_);
      u32x4 g0;
      g0[0] = 1u;                                   // count=1, user mode
      g0[1] = ldsKs;                                // lds_addr
      g0[2] = (unsigned)(ga & 0xffffffffu);         // global_addr[31:0]
      g0[3] = (unsigned)((ga >> 32) & 0x01ffffffu)  // global_addr[56:32]
              | (2u << 30);                         // type=2 (image)
      i32x8 g1;
      g1[0] = 0x00010000;        // wg_mask=0, data_size=1 (2 bytes)
      g1[1] = (int)(64u << 16);  // tensor_dim0 = 64 (lo16 at bit 48)
      g1[2] = (int)(2048u << 16);// tensor_dim1 = 2048 (lo16 at bit 80)
      g1[3] = (int)(64u << 16);  // tile_dim0 = 64 (bits 127:112)
      g1[4] = 32;                // tile_dim1 = 32 (bits 143:128)
      g1[5] = 64;                // tensor_dim0_stride = 64 (bits 207:160)
      g1[6] = 0;
      g1[7] = 0;
      i32x4 z4 = {0, 0, 0, 0};
      i32x8 z8 = {0, 0, 0, 0, 0, 0, 0, 0};
      __builtin_amdgcn_tensor_load_to_lds(g0, g1, z4, z4, z8, 0);
      __builtin_amdgcn_s_wait_tensorcnt(0);
    }
    // --- stage V tile transposed: one b128 load + b16 scatter per thread --
    {
      v8h vv = *(const v8h*)(Vf + base + (size_t)(kv0 + vkv) * D_ + vdb);
      #pragma unroll
      for (int j = 0; j < 8; ++j) Vt[vdb + j][vkv] = vv[j];
    }
    __syncthreads();

    if (kv0 <= q0 + 15) {                     // causal tile skip (wave-uniform)
      v8f s0, s1;
      #pragma unroll
      for (int i = 0; i < 8; ++i) { s0[i] = 0.f; s1[i] = 0.f; }
      #pragma unroll
      for (int c = 0; c < 2; ++c) {
        // B fragment from TDM-staged K tile: contiguous 32B per lane
        v16h kb0 = *(const v16h*)(&Ks[l15][c * 32 + half * 16]);
        v16h kb1 = *(const v16h*)(&Ks[16 + l15][c * 32 + half * 16]);
        s0 = wmma_f16(qf[c].v, kb0, s0);
        s1 = wmma_f16(qf[c].v, kb1, s1);
      }
      float a8[8];
      #pragma unroll
      for (int i = 0; i < 8; ++i) {
        float v0 = s0[i] * 0.125f, v1 = s1[i] * 0.125f;   // 1/sqrt(64)
        const int row = q0 + half * 8 + i;
        if (kv0 +      l15 > row) v0 = -1e30f;
        if (kv0 + 16 + l15 > row) v1 = -1e30f;
        float rmax = fmaxf(v0, v1);
        rmax = fmaxf(rmax, __shfl_xor(rmax, 1, 32));
        rmax = fmaxf(rmax, __shfl_xor(rmax, 2, 32));
        rmax = fmaxf(rmax, __shfl_xor(rmax, 4, 32));
        rmax = fmaxf(rmax, __shfl_xor(rmax, 8, 32));
        const float mn = fmaxf(m8[i], rmax);
        const float alpha = __expf(m8[i] - mn);
        const float p0 = __expf(v0 - mn);
        const float p1 = __expf(v1 - mn);
        float rs = p0 + p1;
        rs += __shfl_xor(rs, 1, 32);
        rs += __shfl_xor(rs, 2, 32);
        rs += __shfl_xor(rs, 4, 32);
        rs += __shfl_xor(rs, 8, 32);
        m8[i] = mn;
        l8[i] = l8[i] * alpha + rs;
        a8[i] = alpha;
        Ps[wave][half * 8 + i][l15]      = (_Float16)p0;
        Ps[wave][half * 8 + i][16 + l15] = (_Float16)p1;
      }
      asm volatile("s_wait_dscnt 0" ::: "memory");  // same-wave LDS RAW
      V16 pf;                                        // P as A-fragment (16x32)
      const _Float16* pr = &Ps[wave][l15][0];
      pf.h[0] = *(const v8h*)(pr + half * 16);
      pf.h[1] = *(const v8h*)(pr + half * 16 + 8);
      #pragma unroll
      for (int c2 = 0; c2 < 4; ++c2) {
        v16h vf = *(const v16h*)(&Vt[c2 * 16 + l15][0] + half * 16);
        #pragma unroll
        for (int i = 0; i < 8; ++i) o[c2][i] *= a8[i];
        o[c2] = wmma_f16(pf.v, vf, o[c2]);
      }
    }
  }

  // Write attn back to [B*T, C] (heads re-interleaved) for proj GEMM.
  #pragma unroll
  for (int c2 = 0; c2 < 4; ++c2) {
    #pragma unroll
    for (int i = 0; i < 8; ++i) {
      const int trow = q0 + half * 8 + i;
      attnO[(size_t)(b * T_ + trow) * C_ + h * D_ + c2 * 16 + l15] = o[c2][i] / l8[i];
    }
  }
}

// ---------------------------------------------------------------------------
extern "C" void kernel_launch(void* const* d_in, const int* in_sizes, int n_in,
                              void* d_out, int out_size, void* d_ws, size_t ws_size,
                              hipStream_t stream) {
  const float* x     = (const float*)d_in[0];
  const float* Wqkv  = (const float*)d_in[1];
  const float* bqkv  = (const float*)d_in[2];
  const float* Wproj = (const float*)d_in[3];
  const float* bproj = (const float*)d_in[4];
  const float* lamb  = (const float*)d_in[5];

  float* out       = (float*)d_out;
  float* value_out = out + (size_t)B_ * T_ * C_;

  // Workspace layout (144 MB):
  //   [0, 96MB)   : qkv fp32 [B*T, 3C]; reused as attn fp32 [B*T, C] after post
  //   [96,112MB)  : Q f16   [B,H,T,D]
  //   [112,128MB) : K f16
  //   [128,144MB) : V f16
  char* ws = (char*)d_ws;
  float*    qkv  = (float*)ws;
  float*    attn = (float*)ws;                              // reuse
  _Float16* Qf   = (_Float16*)(ws + (size_t) 96 * 1024 * 1024);
  _Float16* Kf   = (_Float16*)(ws + (size_t)112 * 1024 * 1024);
  _Float16* Vf   = (_Float16*)(ws + (size_t)128 * 1024 * 1024);

  dim3 blk(256);
  gemm_bias_f16<<<dim3((3 * C_) / 64, (B_ * T_) / 64), blk, 0, stream>>>(
      x, Wqkv, bqkv, qkv, B_ * T_, 3 * C_, C_);
  qkv_post<<<dim3((B_ * H_ * T_) / 8), blk, 0, stream>>>(
      qkv, lamb, value_out, Qf, Kf, Vf);
  flash_attn<<<dim3(T_ / 128, B_ * H_), blk, 0, stream>>>(
      Qf, Kf, Vf, attn);
  gemm_bias_f16<<<dim3(C_ / 64, (B_ * T_) / 64), blk, 0, stream>>>(
      attn, Wproj, bproj, out, B_ * T_, C_, C_);
}
